// CFGEmbeder_83717502534003
// MI455X (gfx1250) — matrix operand order repaired and checked by reading, not compiled, not run on hardware
//
#include <hip/hip_runtime.h>
#include <cstdint>

// Problem constants (match reference)
#define BB 128
#define NN 512
#define HH 512
#define NEG_INF -1e30f

typedef __attribute__((ext_vector_type(16))) _Float16 v16h;
typedef __attribute__((ext_vector_type(8)))  _Float16 v8h;
typedef __attribute__((ext_vector_type(8)))  float    v8f;

// LDS row pitch in halves: 512 + 8 pad -> 1040B stride (16B aligned, and
// 1040/4 = 260 ≡ 4 (mod 64 banks) -> 16 rows read at the same column hit
// 16 disjoint 4-bank groups: conflict-free fragment reads).
#define APITCH 520
#define BPITCH 520
#define USE_ASYNC_LDS 1

// Branch-free tanh: ~8 VALU (v_exp + v_rcp), no EXEC divergence.
// tanh(x) = sign(x) * (1 - e) / (1 + e),  e = exp(-2|x|).
__device__ __forceinline__ float fast_tanh(float x) {
  float e = __expf(-2.0f * __builtin_fabsf(x));
  float r = __fdividef(1.0f - e, 1.0f + e);
  return __builtin_copysignf(r, x);
}

// Raw 32-bit LDS byte offset of a generic pointer into __shared__.
__device__ __forceinline__ uint32_t lds_addr_u32(const void* p) {
  return (uint32_t)(uintptr_t)(__attribute__((address_space(3))) const void*)p;
}

// ---------------------------------------------------------------------------
// Kernel 1: W_sa [H][K] fp32 -> Wt [K][H] f16 (transposed). Transposed rows
// make every WMMA B-fragment / LDS staging chunk a contiguous 16B load.
// ---------------------------------------------------------------------------
__global__ __launch_bounds__(256) void convert_wt_kernel(
    const float* __restrict__ W, _Float16* __restrict__ Wt) {
  int idx = blockIdx.x * 256 + threadIdx.x;   // over HH*HH
  int h = idx & (HH - 1);
  int k = idx >> 9;
  Wt[(size_t)k * HH + h] = (_Float16)W[(size_t)h * HH + k];
}

// ---------------------------------------------------------------------------
// Kernel 2: scores[b,n] = sum_k tanh((X[b,n,:]@W_sa)[k] + b_sa[k]) * w_sc[k] + b_sc
// masked to NEG_INF. grid = B*(N/128); 256 threads = 8 waves; wave = 16-row
// M-tile. A staged+converted to f16 in LDS once. B tile: prologue staged via
// async global->LDS; steady state register-double-buffered (global loads for
// tile j+1 in flight during tile j's 16 WMMAs, then barrier + ds_store).
// Inner loop: 4x ds_load_b128 + v_wmma_f32_16x16x32_f16 (f32 accumulate).
// ---------------------------------------------------------------------------
__global__ __launch_bounds__(256) void scores_kernel(
    const float* __restrict__ X, const int* __restrict__ mask,
    const _Float16* __restrict__ Wt, const float* __restrict__ b_sa,
    const float* __restrict__ w_sc, const float* __restrict__ b_sc,
    float* __restrict__ scores) {
  __shared__ __align__(16) _Float16 Alds[128 * APITCH];  // ~130 KB
  __shared__ __align__(16) _Float16 Blds[16 * BPITCH];   // ~16 KB

  const int b     = blockIdx.x >> 2;
  const int chunk = blockIdx.x & 3;
  const int t     = threadIdx.x;
  const int wave  = t >> 5;
  const int lane  = t & 31;
  const int hsel  = lane >> 4;   // 16-lane half (selects K sub-range)
  const int lcol  = lane & 15;

  // ---- Stage A: 128 rows x 512 K, fp32 -> f16, converted exactly once ----
  const float* Xblk = X + ((size_t)b * NN + chunk * 128) * HH;
#pragma unroll 4
  for (int i = 0; i < 32; ++i) {
    int c   = i * 256 + t;        // 16B-chunk index over 128*64
    int row = c >> 6;
    int cc  = c & 63;
    const float* src = Xblk + (size_t)row * HH + cc * 8;
    float4 f0 = *(const float4*)src;
    float4 f1 = *(const float4*)(src + 4);
    v8h h;
    h[0] = (_Float16)f0.x; h[1] = (_Float16)f0.y;
    h[2] = (_Float16)f0.z; h[3] = (_Float16)f0.w;
    h[4] = (_Float16)f1.x; h[5] = (_Float16)f1.y;
    h[6] = (_Float16)f1.z; h[7] = (_Float16)f1.w;
    *(v8h*)&Alds[row * APITCH + cc * 8] = h;
  }

  // ---- Prologue: stage B tile j=0 via async global->LDS (ASYNCcnt path) ----
  const uint32_t Bbase = lds_addr_u32(Blds);
#pragma unroll
  for (int i = 0; i < 4; ++i) {
    int c  = i * 256 + t;                   // 1024 chunks of 16B
    int r  = c >> 6;
    int cc = c & 63;
    const _Float16* g = Wt + (size_t)r * HH + cc * 8;
#if USE_ASYNC_LDS
    uint32_t loff = Bbase + (uint32_t)(r * BPITCH + cc * 8) * 2u;
    asm volatile("global_load_async_to_lds_b128 %0, %1, off"
                 :: "v"(loff), "v"((uint64_t)(uintptr_t)g)
                 : "memory");
#else
    *(v8h*)&Blds[r * BPITCH + cc * 8] = *(const v8h*)g;
#endif
  }
#if USE_ASYNC_LDS
  asm volatile("s_wait_asynccnt 0x0" ::: "memory");
#endif
  __syncthreads();   // A stage + B tile 0 visible to all waves

  // A layout (16-bit A 16x32): lane holds row (lane&15); halves at
  // k0+hsel*8 (v0-3) and k0+16+hsel*8 (v4-7).
  const int arow = wave * 16 + lcol;
  const _Float16* Arow = &Alds[arow * APITCH];
  // B layout (16-bit B 32x16): lane holds column (lane&15); halves at
  // k0+hsel*8 and k0+16+hsel*8 of transposed-W row (j+lcol).
  const _Float16* Brow = &Blds[lcol * BPITCH];

  float partial[8];
#pragma unroll
  for (int r = 0; r < 8; ++r) partial[r] = 0.f;

  for (int jt = 0; jt < 32; ++jt) {         // 32 output-column tiles of 16
    const int j = jt * 16;
    const bool has_next = (jt + 1) < 32;

    // Register prefetch of next W tile (in flight during the WMMAs below).
    v8h pre[4];
    if (has_next) {
#pragma unroll
      for (int i = 0; i < 4; ++i) {
        int c  = i * 256 + t;
        int r  = c >> 6;
        int cc = c & 63;
        pre[i] = *(const v8h*)(Wt + (size_t)(j + 16 + r) * HH + cc * 8);
      }
    }

    // Epilogue scalars hoisted so WMMAs hide their latency.
    const float bias = b_sa[j + lcol];
    const float wsc  = w_sc[j + lcol];

    v8f c8 = {};
#pragma unroll 4
    for (int k0 = 0; k0 < HH; k0 += 32) {
      v8h a0 = *(const v8h*)(Arow + k0 + hsel * 8);
      v8h a1 = *(const v8h*)(Arow + k0 + 16 + hsel * 8);
      v8h b0 = *(const v8h*)(Brow + k0 + hsel * 8);
      v8h b1 = *(const v8h*)(Brow + k0 + 16 + hsel * 8);
      v16h av = __builtin_shufflevector(a0, a1,
          0,1,2,3,4,5,6,7,8,9,10,11,12,13,14,15);
      v16h bv = __builtin_shufflevector(b0, b1,
          0,1,2,3,4,5,6,7,8,9,10,11,12,13,14,15);
      // (neg_a, A, neg_b, B, c_mod, C, reuse_a, reuse_b)
      c8 = __builtin_amdgcn_wmma_f32_16x16x32_f16(
          false, av, false, bv, (short)0, c8, false, false);
    }

    // C/D layout: (vgpr r, lane) = h[m0 + hsel*8 + r, j + lcol]
#pragma unroll
    for (int r = 0; r < 8; ++r)
      partial[r] += fast_tanh(c8[r] + bias) * wsc;

    if (has_next) {
      __syncthreads();              // all waves done reading current tile
#pragma unroll
      for (int i = 0; i < 4; ++i) {
        int c  = i * 256 + t;
        *(v8h*)&Blds[(c >> 6) * BPITCH + (c & 63) * 8] = pre[i];
      }
      __syncthreads();              // next tile visible
    }
  }

  // Reduce the 16 columns held across lanes of each half (xor<16 keeps bit4).
#pragma unroll
  for (int r = 0; r < 8; ++r) {
    float v = partial[r];
    v += __shfl_xor(v, 1, 32);
    v += __shfl_xor(v, 2, 32);
    v += __shfl_xor(v, 4, 32);
    v += __shfl_xor(v, 8, 32);
    partial[r] = v;
  }

  if (lcol == 0) {
    const float bsc = b_sc[0];
#pragma unroll
    for (int r = 0; r < 8; ++r) {
      int row = chunk * 128 + wave * 16 + hsel * 8 + r;
      float s = partial[r] + bsc;
      size_t idx = (size_t)b * NN + row;
      scores[idx] = mask[idx] ? s : NEG_INF;
    }
  }
}

// ---------------------------------------------------------------------------
// Kernel 3: masked softmax over N per batch + attn-weighted pooling + tanh.
// One block per batch; float2-vectorized, fully coalesced pooling loads.
// ---------------------------------------------------------------------------
__global__ __launch_bounds__(256) void pool_kernel(
    const float* __restrict__ X, const float* __restrict__ scores,
    float* __restrict__ out) {
  __shared__ float attn[NN];
  __shared__ float red_max[8];
  __shared__ float red_sum[8];

  const int b = blockIdx.x;
  const int t = threadIdx.x;

  float s0 = scores[(size_t)b * NN + t];
  float s1 = scores[(size_t)b * NN + t + 256];

  float m = fmaxf(s0, s1);
  m = fmaxf(m, __shfl_xor(m, 1, 32));
  m = fmaxf(m, __shfl_xor(m, 2, 32));
  m = fmaxf(m, __shfl_xor(m, 4, 32));
  m = fmaxf(m, __shfl_xor(m, 8, 32));
  m = fmaxf(m, __shfl_xor(m, 16, 32));
  if ((t & 31) == 0) red_max[t >> 5] = m;
  __syncthreads();
  float M = red_max[0];
#pragma unroll
  for (int i = 1; i < 8; ++i) M = fmaxf(M, red_max[i]);

  float e0 = __expf(s0 - M);   // exp(-1e30 - M) == 0 -> masked weight 0
  float e1 = __expf(s1 - M);
  float sum = e0 + e1;
  sum += __shfl_xor(sum, 1, 32);
  sum += __shfl_xor(sum, 2, 32);
  sum += __shfl_xor(sum, 4, 32);
  sum += __shfl_xor(sum, 8, 32);
  sum += __shfl_xor(sum, 16, 32);
  if ((t & 31) == 0) red_sum[t >> 5] = sum;
  __syncthreads();
  float S = red_sum[0];
#pragma unroll
  for (int i = 1; i < 8; ++i) S += red_sum[i];
  const float inv = 1.f / S;

  attn[t]       = e0 * inv;
  attn[t + 256] = e1 * inv;
  __syncthreads();

  // pooled[h] = sum_n attn[n] * X[b,n,h]; thread t owns h = {2t, 2t+1}
  const float* Xb = X + (size_t)b * NN * HH;
  float acc0 = 0.f, acc1 = 0.f;
#pragma unroll 4
  for (int n = 0; n < NN; ++n) {
    const float a  = attn[n];
    const float2 v = *(const float2*)(Xb + (size_t)n * HH + 2 * t);
    acc0 = fmaf(a, v.x, acc0);
    acc1 = fmaf(a, v.y, acc1);
  }
  float2 o;
  o.x = fast_tanh(acc0);
  o.y = fast_tanh(acc1);
  *(float2*)(out + (size_t)b * HH + 2 * t) = o;
}

// ---------------------------------------------------------------------------
extern "C" void kernel_launch(void* const* d_in, const int* in_sizes, int n_in,
                              void* d_out, int out_size, void* d_ws, size_t ws_size,
                              hipStream_t stream) {
  const float* code_feat = (const float*)d_in[0];   // [B,N,H] fp32
  const int*   node_mask = (const int*)d_in[1];     // [B,N] int32
  const float* W_sa      = (const float*)d_in[2];   // [H,H] fp32
  const float* b_sa      = (const float*)d_in[3];   // [H] fp32
  const float* w_sc      = (const float*)d_in[4];   // [H] fp32
  const float* b_sc      = (const float*)d_in[5];   // scalar fp32
  float*       out       = (float*)d_out;           // [B,H] fp32

  // workspace: [0, H*H*2) = Wt f16 transposed; then scores [B*N] f32
  _Float16* Wt     = (_Float16*)d_ws;
  float*    scores = (float*)((char*)d_ws + (size_t)HH * HH * sizeof(_Float16));

  convert_wt_kernel<<<(HH * HH) / 256, 256, 0, stream>>>(W_sa, Wt);
  scores_kernel<<<BB * (NN / 128), 256, 0, stream>>>(
      code_feat, node_mask, Wt, b_sa, w_sc, b_sc, scores);
  pool_kernel<<<BB, 256, 0, stream>>>(code_feat, scores, out);
}
